// MPSelfAttention_14903536517452
// MI455X (gfx1250) — matrix operand order repaired and checked by reading, not compiled, not run on hardware
//
#include <hip/hip_runtime.h>

#define TOK   3520
#define DIMF  1536
#define NHEAD 12
#define HD    128
#define BLKT  880   // grid_h*grid_w*eff_p = 22*40*1

typedef __attribute__((ext_vector_type(16))) __bf16 v16bf;
typedef __attribute__((ext_vector_type(8)))  __bf16 v8bf;
typedef __attribute__((ext_vector_type(8)))  float  v8f;
typedef __attribute__((ext_vector_type(4)))  float  v4f;

// ---------------------------------------------------------------- helpers --
static __device__ __forceinline__ __bf16 f2bf(float f) {
  union { float f; unsigned u; } c; c.f = f;
  unsigned r = c.u + 0x7FFFu + ((c.u >> 16) & 1u);   // round-to-nearest-even
  union { unsigned short s; __bf16 b; } o; o.s = (unsigned short)(r >> 16);
  return o.b;
}

static __device__ __forceinline__ v8f vzero8() {
  v8f z;
#pragma unroll
  for (int i = 0; i < 8; ++i) z[i] = 0.0f;
  return z;
}

static __device__ __forceinline__ v16bf combine16(v8bf lo, v8bf hi) {
  v16bf r;
#pragma unroll
  for (int i = 0; i < 8; ++i) { r[i] = lo[i]; r[i + 8] = hi[i]; }
  return r;
}

// A operand (16x32 bf16): lane = row m (lane&15), sub = lane>>4.
//   elems 0..7  -> K = base + 8*sub + e
//   elems 8..15 -> K = base + 16 + 8*sub + (e-8)
static __device__ __forceinline__ v16bf load_a_frag(const __bf16* row, int base, int sub) {
  v8bf lo = *(const v8bf*)(row + base + 8 * sub);
  v8bf hi = *(const v8bf*)(row + base + 16 + 8 * sub);
  return combine16(lo, hi);
}

// B operand (32x16 bf16): lane = col n (lane&15), sub = lane>>4.
//   elems 0..7  -> K = base + 8*sub + e
//   elems 8..15 -> K = base + 16 + 8*sub + (e-8)
static __device__ __forceinline__ v16bf load_b_frag(const __bf16* col, int base, int sub) {
  v8bf lo = *(const v8bf*)(col + base + 8 * sub);
  v8bf hi = *(const v8bf*)(col + base + 16 + 8 * sub);
  return combine16(lo, hi);
}

// low 32 bits of a generic LDS pointer = workgroup-relative LDS byte address
static __device__ __forceinline__ unsigned lds_addr(const void* p) {
  return (unsigned)(uintptr_t)p;
}

// async bulk copy: 16 bytes per lane, global -> LDS, tracked by ASYNCcnt
static __device__ __forceinline__ void async_cp16(unsigned lds_off, const void* gptr) {
  asm volatile("global_load_async_to_lds_b128 %0, %1, off"
               :: "v"(lds_off), "v"(gptr) : "memory");
}

// ------------------------------------------------------------ fp32 -> bf16 --
__global__ __launch_bounds__(256) void cvtx_kernel(const float* __restrict__ x,
                                                   __bf16* __restrict__ xh, int n4) {
  int i = blockIdx.x * 256 + threadIdx.x;
  if (i < n4) {
    v4f v = *(const v4f*)(x + (size_t)i * 4);
    __bf16* o = xh + (size_t)i * 4;
    o[0] = f2bf(v[0]); o[1] = f2bf(v[1]); o[2] = f2bf(v[2]); o[3] = f2bf(v[3]);
  }
}

// --------------------------------------- weight transpose + convert (bf16) --
// in:  W  [K=1536][N=1536] fp32 row-major   out: WT [N][K] bf16
__global__ __launch_bounds__(256) void wtrans_kernel(const float* __restrict__ W,
                                                     __bf16* __restrict__ WT) {
  __shared__ float tile[16][17];
  int n0 = blockIdx.x * 16, k0 = blockIdx.y * 16;
  tile[threadIdx.y][threadIdx.x] = W[(size_t)(k0 + threadIdx.y) * DIMF + n0 + threadIdx.x];
  __syncthreads();
  WT[(size_t)(n0 + threadIdx.y) * DIMF + k0 + threadIdx.x] = f2bf(tile[threadIdx.x][threadIdx.y]);
}

// -------------------- bf16 WMMA GEMM, async-LDS double-buffered pipeline ----
// Y[3520][1536] = A[3520][1536] * BT[n][k]^T + bias
// block = 4 waves, 32x32 output tile (2x2 wave-tiles of 16x16).
// per k-chunk (K=32): sA 32x32 bf16 (2KB) + sB 32x32 bf16 (2KB), staged with
// GLOBAL_LOAD_ASYNC_TO_LDS_B128 (each wave issues 1 A-line + 1 B-line = 1KB),
// double buffered so the next chunk is always in flight (ASYNCcnt pipeline).
__global__ __launch_bounds__(128) void gemm_kernel(const __bf16* __restrict__ A,
                                                   const __bf16* __restrict__ BT,
                                                   const float*  __restrict__ bias,
                                                   float* __restrict__ Y) {
  const int lane = threadIdx.x;        // 0..31
  const int wid  = threadIdx.y;        // 0..3
  const int l16  = lane & 15;
  const int sub  = lane >> 4;
  const int wm   = wid >> 1;           // wave-tile row (0/1)
  const int wn   = wid & 1;            // wave-tile col (0/1)
  const int m0   = blockIdx.x * 32;    // 110 row tiles
  const int n0   = blockIdx.y * 32;    // 48 col tiles

  __shared__ __align__(16) __bf16 sA[2][32 * 32];
  __shared__ __align__(16) __bf16 sB[2][32 * 32];

  // async-copy addressing: each instruction moves 8 rows x 64B; wave w covers
  // rows w*8..w*8+7 of both the A chunk and the B chunk.
  const int r8  = lane >> 2;           // 0..7   row within the 8-row slab
  const int c8  = (lane & 3) * 8;      // bf16 column offset (16B granules)
  const int row = wid * 8 + r8;        // 0..31  row within the 32-row chunk
  const unsigned ldsA0 = lds_addr(&sA[0][0]) + (unsigned)(row * 64 + (lane & 3) * 16);
  const unsigned ldsA1 = lds_addr(&sA[1][0]) + (unsigned)(row * 64 + (lane & 3) * 16);
  const unsigned ldsB0 = lds_addr(&sB[0][0]) + (unsigned)(row * 64 + (lane & 3) * 16);
  const unsigned ldsB1 = lds_addr(&sB[1][0]) + (unsigned)(row * 64 + (lane & 3) * 16);
  const __bf16* gA = A  + (size_t)(m0 + row) * DIMF + c8;
  const __bf16* gB = BT + (size_t)(n0 + row) * DIMF + c8;

  // prologue: stage chunk 0 into buffer 0
  async_cp16(ldsA0, gA);
  async_cp16(ldsB0, gB);

  v8f c = vzero8();
  for (int kc = 0; kc < 48; ++kc) {
    const int cur = kc & 1;
    if (kc + 1 < 48) {                 // keep next chunk in flight
      const int koff = (kc + 1) * 32;
      async_cp16(cur ? ldsA0 : ldsA1, gA + koff);
      async_cp16(cur ? ldsB0 : ldsB1, gB + koff);
      asm volatile("s_wait_asynccnt 0x2" ::: "memory");  // current buffer done
    } else {
      asm volatile("s_wait_asynccnt 0x0" ::: "memory");
    }
    __syncthreads();                   // all waves' slabs visible

    const __bf16* pa = &sA[cur][(wm * 16 + l16) * 32];
    const __bf16* pb = &sB[cur][(wn * 16 + l16) * 32];
    v16bf a = load_a_frag(pa, 0, sub);
    v16bf b = load_b_frag(pb, 0, sub);
    c = __builtin_amdgcn_wmma_f32_16x16x32_bf16(false, a, false, b, (short)0, c, false, false);

    __syncthreads();                   // done reading before next overwrite
  }

  const float bv = bias[n0 + wn * 16 + l16];
#pragma unroll
  for (int r = 0; r < 8; ++r) {
    int rr = m0 + wm * 16 + r + 8 * sub;   // C layout: lanes 16-31 hold M=r+8
    Y[(size_t)rr * DIMF + n0 + wn * 16 + l16] = c[r] + bv;
  }
}

// -------------------------------- fused RMS-norm + RoPE + V transpose ------
__global__ __launch_bounds__(256) void normrope_kernel(
    const float* __restrict__ qlin, const float* __restrict__ klin,
    const float* __restrict__ vlin, const float* __restrict__ nqw,
    const float* __restrict__ nkw,  const float* __restrict__ freqs,
    __bf16* __restrict__ qbf, __bf16* __restrict__ kbf, __bf16* __restrict__ vtb) {
  const int t   = blockIdx.x;
  const int tid = threadIdx.x;
  __shared__ float red[2][256];

  const float* qrow = qlin + (size_t)t * DIMF;
  const float* krow = klin + (size_t)t * DIMF;
  float sq = 0.f, sk = 0.f;
  for (int i = tid; i < DIMF; i += 256) {
    float a = qrow[i]; sq += a * a;
    float b = krow[i]; sk += b * b;
  }
  red[0][tid] = sq; red[1][tid] = sk;
  __syncthreads();
  for (int s = 128; s > 0; s >>= 1) {
    if (tid < s) { red[0][tid] += red[0][tid + s]; red[1][tid] += red[1][tid + s]; }
    __syncthreads();
  }
  const float rq = rsqrtf(red[0][0] * (1.0f / DIMF) + 1e-6f);
  const float rk = rsqrtf(red[1][0] * (1.0f / DIMF) + 1e-6f);

  const int fi = t / BLKT, hi = (t % BLKT) / 40, wi = t % 40;
#pragma unroll
  for (int pp = 0; pp < 3; ++pp) {                    // 768 rope pairs / token
    int p = tid + pp * 256;
    int head = p >> 6, j = p & 63;
    int pos = (j < 22) ? fi : ((j < 43) ? hi : wi);   // temporal/h/w freq split
    float cs = freqs[(size_t)pos * 128 + 2 * j];
    float sn = freqs[(size_t)pos * 128 + 2 * j + 1];
    int base = head * 128 + 2 * j;
    float qr = qrow[base]     * rq * nqw[base];
    float qi = qrow[base + 1] * rq * nqw[base + 1];
    float kr = krow[base]     * rk * nkw[base];
    float ki = krow[base + 1] * rk * nkw[base + 1];
    size_t ob = ((size_t)(head * TOK + t)) * HD + 2 * j;
    qbf[ob]     = f2bf(qr * cs - qi * sn);
    qbf[ob + 1] = f2bf(qr * sn + qi * cs);
    kbf[ob]     = f2bf(kr * cs - ki * sn);
    kbf[ob + 1] = f2bf(kr * sn + ki * cs);
  }
  const float* vrow = vlin + (size_t)t * DIMF;
  for (int i = tid; i < DIMF; i += 256) {             // V -> [H][128][T]
    int head = i >> 7, d = i & 127;
    vtb[((size_t)(head * HD + d)) * TOK + t] = f2bf(vrow[i]);
  }
}

// ------------------------------------- flash attention, 1 wave / 16 queries --
__global__ __launch_bounds__(32) void attn_kernel(const __bf16* __restrict__ qbf,
                                                  const __bf16* __restrict__ kbf,
                                                  const __bf16* __restrict__ vtb,
                                                  __bf16* __restrict__ aout) {
  const int h    = blockIdx.x;        // 12 heads
  const int qt   = blockIdx.y;        // 220 query tiles
  const int lane = threadIdx.x;
  const int l16  = lane & 15;
  const int sub  = lane >> 4;

  __shared__ __align__(16) __bf16 Plds[16 * 32];      // P tile, C->A transpose

  // Q fragments for the 4 K-chunks of the 128-dim head
  const __bf16* qp = qbf + ((size_t)(h * TOK + qt * 16 + l16)) * HD;
  v16bf aq[4];
#pragma unroll
  for (int kc = 0; kc < 4; ++kc) aq[kc] = load_a_frag(qp, kc * 32, sub);

  v8f acc[8];
#pragma unroll
  for (int n = 0; n < 8; ++n) acc[n] = vzero8();
  float mrun[8], lrun[8];
#pragma unroll
  for (int r = 0; r < 8; ++r) { mrun[r] = -3.0e38f; lrun[r] = 0.0f; }

  const int   limit = ((qt * 16) / BLKT + 1) * BLKT;  // block-causal key bound
  const int   nc    = (limit + 31) / 32;
  const float scale = 0.08838834764831845f;           // 1/sqrt(128)

  for (int c = 0; c < nc; ++c) {
    const int kb = c * 32;
    // ---- S = Q K^T for 32 keys (two 16-key column groups) ----
    v8f s0 = vzero8(), s1 = vzero8();
    const __bf16* kp0 = kbf + ((size_t)(h * TOK + kb + l16)) * HD;
    const __bf16* kp1 = kbf + ((size_t)(h * TOK + kb + 16 + l16)) * HD;
#pragma unroll
    for (int kc = 0; kc < 4; ++kc) {
      v16bf b0 = load_b_frag(kp0, kc * 32, sub);
      s0 = __builtin_amdgcn_wmma_f32_16x16x32_bf16(false, aq[kc], false, b0, (short)0, s0, false, false);
      v16bf b1 = load_b_frag(kp1, kc * 32, sub);
      s1 = __builtin_amdgcn_wmma_f32_16x16x32_bf16(false, aq[kc], false, b1, (short)0, s1, false, false);
    }
    const bool msk0 = (kb + l16) >= limit;
    const bool msk1 = (kb + 16 + l16) >= limit;
    float sa[8], sb[8];
#pragma unroll
    for (int r = 0; r < 8; ++r) {
      sa[r] = msk0 ? -3.0e38f : s0[r] * scale;        // branchless column mask
      sb[r] = msk1 ? -3.0e38f : s1[r] * scale;
    }
    // ---- online softmax (rows of a half live in lanes 0-15 / 16-31) ----
#pragma unroll
    for (int r = 0; r < 8; ++r) {
      float cm = fmaxf(sa[r], sb[r]);
      cm = fmaxf(cm, __shfl_xor(cm, 1, 32));
      cm = fmaxf(cm, __shfl_xor(cm, 2, 32));
      cm = fmaxf(cm, __shfl_xor(cm, 4, 32));
      cm = fmaxf(cm, __shfl_xor(cm, 8, 32));
      float mnew  = fmaxf(mrun[r], cm);
      float alpha = __expf(mrun[r] - mnew);
      float p0 = __expf(sa[r] - mnew);
      float p1 = __expf(sb[r] - mnew);
      float rs = p0 + p1;
      rs += __shfl_xor(rs, 1, 32);
      rs += __shfl_xor(rs, 2, 32);
      rs += __shfl_xor(rs, 4, 32);
      rs += __shfl_xor(rs, 8, 32);
      lrun[r] = lrun[r] * alpha + rs;
      mrun[r] = mnew;
#pragma unroll
      for (int n = 0; n < 8; ++n) acc[n][r] *= alpha;
      int row = r + 8 * sub;                          // C layout -> LDS row-major
      Plds[row * 32 + l16]      = f2bf(p0);
      Plds[row * 32 + 16 + l16] = f2bf(p1);
    }
    __syncthreads();
    const __bf16* pl = Plds + l16 * 32;               // lane's A row is l16
    v16bf ap = load_a_frag(pl, 0, sub);
    __syncthreads();
    // ---- O += P V  (8 column tiles of 16 over the 128 head dims) ----
#pragma unroll
    for (int n = 0; n < 8; ++n) {
      const __bf16* vpn = vtb + ((size_t)(h * HD + n * 16 + l16)) * TOK + kb;
      v16bf bv = load_b_frag(vpn, 0, sub);
      acc[n] = __builtin_amdgcn_wmma_f32_16x16x32_bf16(false, ap, false, bv, (short)0, acc[n], false, false);
    }
  }
  // ---- epilogue: normalize and pack heads back to [T][1536] bf16 ----
#pragma unroll
  for (int r = 0; r < 8; ++r) {
    int   row   = r + 8 * sub;
    int   token = qt * 16 + row;
    float inv   = 1.0f / lrun[r];
#pragma unroll
    for (int n = 0; n < 8; ++n)
      aout[(size_t)token * DIMF + h * HD + n * 16 + l16] = f2bf(acc[n][r] * inv);
  }
}

// ----------------------------------------------------------------- launch --
extern "C" void kernel_launch(void* const* d_in, const int* in_sizes, int n_in,
                              void* d_out, int out_size, void* d_ws, size_t ws_size,
                              hipStream_t stream) {
  const float* x   = (const float*)d_in[0];
  const float* q_w = (const float*)d_in[1];
  const float* q_b = (const float*)d_in[2];
  const float* k_w = (const float*)d_in[3];
  const float* k_b = (const float*)d_in[4];
  const float* v_w = (const float*)d_in[5];
  const float* v_b = (const float*)d_in[6];
  const float* o_w = (const float*)d_in[7];
  const float* o_b = (const float*)d_in[8];
  const float* nqw = (const float*)d_in[9];
  const float* nkw = (const float*)d_in[10];
  const float* fr  = (const float*)d_in[11];

  char*  ws  = (char*)d_ws;
  size_t off = 0;
  __bf16* xh   = (__bf16*)(ws + off); off += (size_t)TOK * DIMF * 2;
  __bf16* wqT  = (__bf16*)(ws + off); off += (size_t)DIMF * DIMF * 2;
  __bf16* wkT  = (__bf16*)(ws + off); off += (size_t)DIMF * DIMF * 2;
  __bf16* wvT  = (__bf16*)(ws + off); off += (size_t)DIMF * DIMF * 2;
  __bf16* woT  = (__bf16*)(ws + off); off += (size_t)DIMF * DIMF * 2;
  float*  qlin = (float*)(ws + off);  off += (size_t)TOK * DIMF * 4;
  float*  klin = (float*)(ws + off);  off += (size_t)TOK * DIMF * 4;
  float*  vlin = (float*)(ws + off);  off += (size_t)TOK * DIMF * 4;
  __bf16* qbf  = (__bf16*)(ws + off); off += (size_t)TOK * DIMF * 2;
  __bf16* kbf  = (__bf16*)(ws + off); off += (size_t)TOK * DIMF * 2;
  __bf16* vtb  = (__bf16*)(ws + off); off += (size_t)TOK * DIMF * 2;
  __bf16* aout = (__bf16*)(ws + off); off += (size_t)TOK * DIMF * 2;

  // 1) x -> bf16
  { int n4 = TOK * DIMF / 4;
    cvtx_kernel<<<(n4 + 255) / 256, 256, 0, stream>>>(x, xh, n4); }

  // 2) weight transpose+convert
  { dim3 tb(16, 16), tg(DIMF / 16, DIMF / 16);
    wtrans_kernel<<<tg, tb, 0, stream>>>(q_w, wqT);
    wtrans_kernel<<<tg, tb, 0, stream>>>(k_w, wkT);
    wtrans_kernel<<<tg, tb, 0, stream>>>(v_w, wvT);
    wtrans_kernel<<<tg, tb, 0, stream>>>(o_w, woT); }

  // 3) QKV projections (bf16 WMMA, async-LDS pipelined)
  dim3 gb(32, 4), gg(TOK / 32, DIMF / 32);
  gemm_kernel<<<gg, gb, 0, stream>>>(xh, wqT, q_b, qlin);
  gemm_kernel<<<gg, gb, 0, stream>>>(xh, wkT, k_b, klin);
  gemm_kernel<<<gg, gb, 0, stream>>>(xh, wvT, v_b, vlin);

  // 4) RMS-norm + RoPE + V transpose
  normrope_kernel<<<TOK, 256, 0, stream>>>(qlin, klin, vlin, nqw, nkw, fr, qbf, kbf, vtb);

  // 5) block-causal flash attention (bf16 WMMA)
  attn_kernel<<<dim3(NHEAD, TOK / 16), 32, 0, stream>>>(qbf, kbf, vtb, aout);

  // 6) output projection -> fp32 d_out
  gemm_kernel<<<gg, gb, 0, stream>>>(aout, woT, o_b, (float*)d_out);
}
